// ImplicitField_32804960207257
// MI455X (gfx1250) — matrix operand order repaired and checked by reading, not compiled, not run on hardware
//
#include <hip/hip_runtime.h>
#include <stdint.h>

typedef __attribute__((ext_vector_type(16))) _Float16 v16h;
typedef __attribute__((ext_vector_type(8)))  float    v8f;

#define GDIM 64
constexpr int Bc    = 4;
constexpr int Cc    = 32;
constexpr int Nc    = 131072;          // points per batch (2^17)
constexpr int INDIM = 35;              // C + 3
constexpr int PTS   = Bc * Nc;         // 524288 total points
constexpr int WAVES = 8;               // waves per workgroup
constexpr int G3    = GDIM * GDIM * GDIM;

union FragH { v16h h; uint32_t u[8]; uint4 q[2]; };

__device__ inline uint32_t pack_h2(float a, float b) {
  union { _Float16 h[2]; uint32_t u; } p;
  p.h[0] = (_Float16)a; p.h[1] = (_Float16)b;
  return p.u;
}

// ---------------------------------------------------------------------------
// Prep: swizzle w1 (128x35, K-padded to 64) and w2 (128x128) into WMMA
// B-matrix fragments (f16). Fragment f stored as 256 dwords: lane l owns
// dwords [l*8 .. l*8+7]; dword v of lane l = packed f16 pair
// (K = (l>>4)*16 + 2v, N = nt*16 + (l&15)). 16 frags for w1, 32 for w2.
// ---------------------------------------------------------------------------
__global__ __launch_bounds__(256) void prep_frags(
    const float* __restrict__ w1, const float* __restrict__ w2,
    uint32_t* __restrict__ frags) {
  int idx = blockIdx.x * 256 + threadIdx.x;
  if (idx >= 48 * 256) return;
  int f = idx >> 8;          // fragment index
  int r = idx & 255;
  int l = r >> 3, v = r & 7;
  int ncol = l & 15, khalf = l >> 4;
  uint32_t pack;
  if (f < 16) {              // w1: kt in [0,2), nt in [0,8)
    int kt = f >> 3, nt = f & 7;
    int n = nt * 16 + ncol;
    int k = kt * 32 + khalf * 16 + v * 2;
    float f0 = (k     < INDIM) ? w1[n * INDIM + k]     : 0.0f;
    float f1 = (k + 1 < INDIM) ? w1[n * INDIM + k + 1] : 0.0f;
    pack = pack_h2(f0, f1);
  } else {                   // w2: kt in [0,4), nt in [0,8)
    int f2 = f - 16;
    int kt = f2 >> 3, nt = f2 & 7;
    int n = nt * 16 + ncol;
    int k = kt * 32 + khalf * 16 + v * 2;
    pack = pack_h2(w2[n * 128 + k], w2[n * 128 + k + 1]);
  }
  frags[idx] = pack;
}

// ---------------------------------------------------------------------------
// Fused: trilinear sample -> [35->128 relu] -> [128->128 relu] -> [128->1]
// One wave handles 16 points (M=16 tile) through the whole pipeline.
// ---------------------------------------------------------------------------
__global__ __launch_bounds__(WAVES * 32) void field_fused(
    const float* __restrict__ feat, const float* __restrict__ pts,
    const float* __restrict__ b1,   const float* __restrict__ b2,
    const float* __restrict__ w3,   const float* __restrict__ b3,
    const uint32_t* __restrict__ frag1, const uint32_t* __restrict__ frag2,
    float* __restrict__ out) {
  // Per-wave 4KB slab: 16 rows x 128 f16 (x-tile uses K=0..63; H1 uses 0..127)
  __shared__ uint32_t lds32[WAVES][16 * 64];

  const int wave  = threadIdx.x >> 5;
  const int l     = threadIdx.x & 31;
  const int row   = l & 15;          // point row in tile / D-layout column sel
  const int khalf = l >> 4;

  // Warm the weight-fragment lines (gfx1250 global_prefetch path).
  __builtin_prefetch(frag2 + (size_t)l * 256, 0, 1);

  const int tile = blockIdx.x * WAVES + wave;
  const int pidx = tile * 16 + row;
  const int bb   = pidx >> 17;       // Nc == 2^17

  // ---- trilinear sampling (reference's double-normalization kept) ----
  const float* pp = pts + (size_t)pidx * 3;
  float px = pp[0], py = pp[1], pz = pp[2];
  // pixel coord = (((p+1)/2)+1)/2 * 63 = (p+3) * 63/4
  float fx = (px + 3.0f) * (63.0f * 0.25f);
  float fy = (py + 3.0f) * (63.0f * 0.25f);
  float fz = (pz + 3.0f) * (63.0f * 0.25f);
  float fx0 = floorf(fx), fy0 = floorf(fy), fz0 = floorf(fz);
  float wx = fx - fx0, wy = fy - fy0, wz = fz - fz0;
  int x0 = min(max((int)fx0, 0), GDIM - 1); int x1 = min(x0 + 1, GDIM - 1);
  int y0 = min(max((int)fy0, 0), GDIM - 1); int y1 = min(y0 + 1, GDIM - 1);
  int z0 = min(max((int)fz0, 0), GDIM - 1); int z1 = min(z0 + 1, GDIM - 1);
  int zy00 = z0 * 4096 + y0 * 64, zy01 = z0 * 4096 + y1 * 64;
  int zy10 = z1 * 4096 + y0 * 64, zy11 = z1 * 4096 + y1 * 64;
  int o000 = zy00 + x0, o001 = zy00 + x1, o010 = zy01 + x0, o011 = zy01 + x1;
  int o100 = zy10 + x0, o101 = zy10 + x1, o110 = zy11 + x0, o111 = zy11 + x1;
  float u0 = 1.0f - wx, u1 = wx, v0 = 1.0f - wy, v1 = wy, s0 = 1.0f - wz, s1 = wz;
  float w000 = s0 * v0 * u0, w001 = s0 * v0 * u1, w010 = s0 * v1 * u0, w011 = s0 * v1 * u1;
  float w100 = s1 * v0 * u0, w101 = s1 * v0 * u1, w110 = s1 * v1 * u0, w111 = s1 * v1 * u1;

  _Float16* xt = (_Float16*)&lds32[wave][0];
  const size_t fb = (size_t)bb * Cc * G3;
#pragma unroll
  for (int i = 0; i < 16; ++i) {
    int c = khalf * 16 + i;
    const float* fc = feat + fb + (size_t)c * G3;
    float v = w000 * fc[o000] + w001 * fc[o001] + w010 * fc[o010] + w011 * fc[o011]
            + w100 * fc[o100] + w101 * fc[o101] + w110 * fc[o110] + w111 * fc[o111];
    xt[row * 128 + 3 + c] = (_Float16)v;
  }
  if (khalf == 0) {
    xt[row * 128 + 0] = (_Float16)px;
    xt[row * 128 + 1] = (_Float16)py;
    xt[row * 128 + 2] = (_Float16)pz;
#pragma unroll
    for (int k = INDIM; k < 64; ++k) xt[row * 128 + k] = (_Float16)0.0f;
    // pass-through: out[0 .. PTS*3) = points
    out[(size_t)pidx * 3 + 0] = px;
    out[(size_t)pidx * 3 + 1] = py;
    out[(size_t)pidx * 3 + 2] = pz;
  }
  asm volatile("s_wait_dscnt 0x0" ::: "memory");

  // ---- layer 1: H1[16x128] = X[16x64] * W1^T, relu ----
  // A-frag layout (16-bit A 16x32): lane (row,khalf), VGPR v -> K =
  //   kt*32 + khalf*16 + v*2
  FragH a1[2];
#pragma unroll
  for (int kt = 0; kt < 2; ++kt)
#pragma unroll
    for (int v = 0; v < 8; ++v) {
      int kb = kt * 32 + khalf * 16 + v * 2;
      a1[kt].u[v] = lds32[wave][row * 64 + (kb >> 1)];
    }
#pragma unroll
  for (int nt = 0; nt < 8; ++nt) {
    v8f acc = {};
#pragma unroll
    for (int kt = 0; kt < 2; ++kt) {
      FragH bf;
      const uint4* bp = (const uint4*)(frag1 + ((kt * 8 + nt) * 256 + l * 8));
      bf.q[0] = bp[0]; bf.q[1] = bp[1];
      acc = __builtin_amdgcn_wmma_f32_16x16x32_f16(
          false, a1[kt].h, false, bf.h, (short)0, acc, false, false);
    }
    int col = nt * 16 + row;           // D layout: lane owns one column
    float bias = b1[col];
#pragma unroll
    for (int r = 0; r < 8; ++r) {
      float hv = acc[r] + bias;
      hv = hv > 0.0f ? hv : 0.0f;
      xt[(r + khalf * 8) * 128 + col] = (_Float16)hv;   // restage H1 f16
    }
  }
  asm volatile("s_wait_dscnt 0x0" ::: "memory");

  // ---- layer 2 (+ fused layer 3): H2 = relu(H1*W2^T + b2); d = H2*w3 ----
  FragH a2[4];
#pragma unroll
  for (int kt = 0; kt < 4; ++kt)
#pragma unroll
    for (int v = 0; v < 8; ++v) {
      int kb = kt * 32 + khalf * 16 + v * 2;
      a2[kt].u[v] = lds32[wave][row * 64 + (kb >> 1)];
    }
  float dacc[8];
#pragma unroll
  for (int r = 0; r < 8; ++r) dacc[r] = 0.0f;
#pragma unroll
  for (int nt = 0; nt < 8; ++nt) {
    v8f acc = {};
#pragma unroll
    for (int kt = 0; kt < 4; ++kt) {
      FragH bf;
      const uint4* bp = (const uint4*)(frag2 + ((kt * 8 + nt) * 256 + l * 8));
      bf.q[0] = bp[0]; bf.q[1] = bp[1];
      acc = __builtin_amdgcn_wmma_f32_16x16x32_f16(
          false, a2[kt].h, false, bf.h, (short)0, acc, false, false);
    }
    int col = nt * 16 + row;
    float bias = b2[col];
    float w3v  = w3[col];
#pragma unroll
    for (int r = 0; r < 8; ++r) {
      float hv = acc[r] + bias;
      hv = hv > 0.0f ? hv : 0.0f;
      dacc[r] += hv * w3v;             // layer-3 partial (this lane's columns)
    }
  }

  // ---- layer-3 reduction across the 16 column-lanes via LDS ----
  float* red = (float*)&lds32[wave][0];      // 16 rows x 16 partials
#pragma unroll
  for (int r = 0; r < 8; ++r)
    red[(r + khalf * 8) * 16 + row] = dacc[r];
  asm volatile("s_wait_dscnt 0x0" ::: "memory");
  if (l < 16) {
    float s = b3[0];
#pragma unroll
    for (int j = 0; j < 16; ++j) s += red[l * 16 + j];
    out[(size_t)PTS * 3 + tile * 16 + l] = s;   // densities after points
  }
}

// ---------------------------------------------------------------------------
extern "C" void kernel_launch(void* const* d_in, const int* in_sizes, int n_in,
                              void* d_out, int out_size, void* d_ws, size_t ws_size,
                              hipStream_t stream) {
  const float* feat = (const float*)d_in[0];
  const float* pts  = (const float*)d_in[1];
  const float* w1   = (const float*)d_in[2];
  const float* b1   = (const float*)d_in[3];
  const float* w2   = (const float*)d_in[4];
  const float* b2   = (const float*)d_in[5];
  const float* w3   = (const float*)d_in[6];
  const float* b3   = (const float*)d_in[7];

  uint32_t* frags = (uint32_t*)d_ws;           // 48 frags * 1KB = 48KB
  prep_frags<<<48, 256, 0, stream>>>(w1, w2, frags);

  const int blocks = PTS / (16 * WAVES);       // 4096
  field_fused<<<blocks, WAVES * 32, 0, stream>>>(
      feat, pts, b1, b2, w3, b3,
      frags, frags + 16 * 256, (float*)d_out);
}
